// VectorQuantizer_28415503630708
// MI455X (gfx1250) — matrix operand order repaired and checked by reading, not compiled, not run on hardware
//
#include <hip/hip_runtime.h>
#include <hip/hip_bf16.h>
#include <stdint.h>

// ---------------------------------------------------------------------------
// VQ-VAE vector quantizer for gfx1250 (MI455X), wave32 + WMMA bf16 split-f32,
// with ASYNC global->LDS staging of the codebook (ASYNCcnt / s_wait_asynccnt).
// x:   [32,128,32,32] f32   -> per batch b, X_b is a [128 x 1024] matrix with
//                              contiguous columns (latent vectors).
// emb: [512,128] f32
// out: [quantized 4194304][vq_loss 1][indices 32768][usage 1]  (all f32)
// ---------------------------------------------------------------------------

typedef __attribute__((ext_vector_type(16))) __bf16 v16bf;
typedef __attribute__((ext_vector_type(8)))  float  v8f;

#define C_DIM   128
#define K_CODES 512
#define HW      1024
#define COLS_PER_BLOCK 128
#define THREADS 256
#define KCHUNK  64
#define NCHUNKS 8          // 512 / 64
#define EPITCH  136        // LDS row pitch (bf16 elems): 272B, 16B aligned, bank-spread

#define Q_ELEMS  4194304   // 32*128*32*32
#define N_TOTAL  32768

struct __align__(16) U4 { unsigned a, b, c, d; };
union AFrag { v16bf v; U4 q[2]; };
union BFrag { v16bf v; unsigned u[8]; };

__device__ __forceinline__ unsigned short f32_to_bf16_rn(float f) {
    unsigned u = __builtin_bit_cast(unsigned, f);
    unsigned r = u + 0x7FFFu + ((u >> 16) & 1u);     // round-to-nearest-even
    return (unsigned short)(r >> 16);
}
__device__ __forceinline__ float bf16_to_f32(unsigned short h) {
    unsigned u = ((unsigned)h) << 16;
    return __builtin_bit_cast(float, u);
}

// LDS byte-offset of a __shared__ object (addrspace(3) pointers are 32-bit
// offsets from LDS_BASE, which is what GLOBAL_LOAD_ASYNC_TO_LDS's VDST wants).
__device__ __forceinline__ unsigned lds_addr_of(const void* p) {
    return (unsigned)(unsigned long long)(const __attribute__((address_space(3))) void*)p;
}

// ---------------------------------------------------------------------------
// Kernel 1: split emb into bf16 hi/lo, code norms, zero usage flags.
// ---------------------------------------------------------------------------
__global__ void vq_prep(const float* __restrict__ emb,
                        unsigned short* __restrict__ ehi,
                        unsigned short* __restrict__ elo,
                        float* __restrict__ enorm,
                        float* __restrict__ used) {
    int g = blockIdx.x * blockDim.x + threadIdx.x;
    if (g < K_CODES * C_DIM) {
        float f = emb[g];
        unsigned short h = f32_to_bf16_rn(f);
        ehi[g] = h;
        elo[g] = f32_to_bf16_rn(f - bf16_to_f32(h));
    }
    if (g < K_CODES) {
        float s = 0.f;
        const float* row = emb + g * C_DIM;
#pragma unroll 8
        for (int c = 0; c < C_DIM; ++c) { float v = row[c]; s += v * v; }
        enorm[g] = s;
        used[g]  = 0.f;
    }
}

// ---------------------------------------------------------------------------
// Kernel 2: per-block: 128 columns; 8 waves x 16 cols each.
// Scores via D = emb_tile(16x32) x X_tile(32x16) WMMA, split bf16 (3 passes).
// Codebook chunks staged with global_load_async_to_lds_b128 (ASYNCcnt).
// ---------------------------------------------------------------------------
__global__ void __launch_bounds__(THREADS)
vq_main(const float* __restrict__ x,
        const float* __restrict__ emb,
        const unsigned short* __restrict__ g_ehi,
        const unsigned short* __restrict__ g_elo,
        const float* __restrict__ g_enorm,
        float* __restrict__ used,
        float* __restrict__ partials,
        float* __restrict__ out_q,
        float* __restrict__ out_idx) {

    __shared__ union {
        struct { unsigned short ehi[KCHUNK * EPITCH];
                 unsigned short elo[KCHUNK * EPITCH]; } es;
    } smem;
    __shared__ float enorm_s[K_CODES];
    __shared__ int   idx_s[COLS_PER_BLOCK];
    __shared__ float red_s[THREADS];

    const int tid  = threadIdx.x;
    const int bb   = blockIdx.x;
    const int n0   = bb * COLS_PER_BLOCK;       // first global column
    const int b    = n0 >> 10;                  // batch
    const int hw0  = n0 & 1023;                 // first hw within batch
    const float* xb = x + ((size_t)b * C_DIM) * HW + hw0;  // xb[c*HW + i]

    for (int k = tid; k < K_CODES; k += THREADS) enorm_s[k] = g_enorm[k];

    const int lane = tid & 31;
    const int wave = tid >> 5;
    const int hl   = lane >> 4;                 // lane-half (0/1)
    const int nl   = (wave << 4) + (lane & 15); // this lane's column (in block)

    const unsigned lds_ehi = lds_addr_of(&smem.es.ehi[0]);
    const unsigned lds_elo = lds_addr_of(&smem.es.elo[0]);

    // --- B fragments, loaded straight from global (columns are contiguous
    //     across lanes -> coalesced), split into bf16 hi/lo; kept in VGPRs
    //     for the entire kernel. Layout: lanes 0-15 hold K=0..15 (2/VGPR),
    //     lanes 16-31 hold K=16..31.
    BFrag bhi[4], blo[4];
    {
        const float* xcol = xb + nl;
#pragma unroll
        for (int s = 0; s < 4; ++s) {
#pragma unroll
            for (int v = 0; v < 8; ++v) {
                int c = s * 32 + hl * 16 + 2 * v;
                float f0 = xcol[(size_t)c * HW];
                float f1 = xcol[(size_t)(c + 1) * HW];
                unsigned short h0 = f32_to_bf16_rn(f0);
                unsigned short h1 = f32_to_bf16_rn(f1);
                unsigned short l0 = f32_to_bf16_rn(f0 - bf16_to_f32(h0));
                unsigned short l1 = f32_to_bf16_rn(f1 - bf16_to_f32(h1));
                bhi[s].u[v] = (unsigned)h0 | ((unsigned)h1 << 16);
                blo[s].u[v] = (unsigned)l0 | ((unsigned)l1 << 16);
            }
        }
    }

    float best = 3.0e38f;
    int   bidx = 0;

    for (int ch = 0; ch < NCHUNKS; ++ch) {
        __syncthreads();
        // ---- async stage of 64 codes (hi+lo) into padded LDS: 256 B of
        // contiguous global per row -> 272B-pitch LDS rows, 16B granules.
        {
            const void* ghi = (const void*)(g_ehi + (size_t)(ch * KCHUNK) * C_DIM);
            const void* glo = (const void*)(g_elo + (size_t)(ch * KCHUNK) * C_DIM);
#pragma unroll
            for (int it = 0; it < (KCHUNK * 16) / THREADS; ++it) {
                int e = tid + it * THREADS;          // 16B granule id
                int r = e >> 4, cq = e & 15;
                unsigned dlhi = lds_ehi + (unsigned)(r * (EPITCH * 2) + cq * 16);
                unsigned dllo = lds_elo + (unsigned)(r * (EPITCH * 2) + cq * 16);
                unsigned go   = (unsigned)(r * 256 + cq * 16);
                asm volatile("global_load_async_to_lds_b128 %0, %1, %2"
                             :: "v"(dlhi), "v"(go), "s"(ghi) : "memory");
                asm volatile("global_load_async_to_lds_b128 %0, %1, %2"
                             :: "v"(dllo), "v"(go), "s"(glo) : "memory");
            }
        }
        asm volatile("s_wait_asynccnt 0" ::: "memory");
        __syncthreads();

#pragma unroll
        for (int t = 0; t < 4; ++t) {
            const int row = t * 16 + (lane & 15);   // A-matrix row = code-in-tile
            v8f acc = {0.f, 0.f, 0.f, 0.f, 0.f, 0.f, 0.f, 0.f};
#pragma unroll
            for (int s = 0; s < 4; ++s) {
                // A layout: lanes 0-15: K=0..15 (v0-7);
                //           lanes 16-31: K=16..31  -> two b128 loads.
                AFrag ahi, alo;
                const int c0 = s * 32 + hl * 16;
                ahi.q[0] = *(const U4*)&smem.es.ehi[row * EPITCH + c0];
                ahi.q[1] = *(const U4*)&smem.es.ehi[row * EPITCH + c0 + 8];
                alo.q[0] = *(const U4*)&smem.es.elo[row * EPITCH + c0];
                alo.q[1] = *(const U4*)&smem.es.elo[row * EPITCH + c0 + 8];
                // split-f32: hi*hi + hi*lo + lo*hi  (lo*lo negligible)
                acc = __builtin_amdgcn_wmma_f32_16x16x32_bf16(false, ahi.v, false, bhi[s].v,
                                                              (short)0, acc, false, false);
                acc = __builtin_amdgcn_wmma_f32_16x16x32_bf16(false, ahi.v, false, blo[s].v,
                                                              (short)0, acc, false, false);
                acc = __builtin_amdgcn_wmma_f32_16x16x32_bf16(false, alo.v, false, bhi[s].v,
                                                              (short)0, acc, false, false);
            }
            // D layout: VGPR j holds code M = j + 8*hl, column N = lane&15
            const int cb = ch * KCHUNK + t * 16 + hl * 8;
#pragma unroll
            for (int j = 0; j < 8; ++j) {
                float sc = enorm_s[cb + j] - 2.0f * acc[j];
                if (sc < best) { best = sc; bidx = cb + j; }
            }
        }
    }

    // combine the two lane-halves (same column, complementary code rows)
    {
        float ob = __shfl_xor(best, 16, 32);
        int   oi = __shfl_xor(bidx, 16, 32);
        if (ob < best || (ob == best && oi < bidx)) { best = ob; bidx = oi; }
    }

    if (lane < 16) {
        int ncol = (wave << 4) + lane;
        idx_s[ncol] = bidx;
        out_idx[n0 + ncol] = (float)bidx;
        used[bidx] = 1.0f;      // benign same-value race, deterministic
    }
    __syncthreads();

    // phase 2: write quantized = emb[idx] back in (B,C,H,W) layout + loss
    float lacc = 0.f;
    for (int e = tid; e < C_DIM * COLS_PER_BLOCK; e += THREADS) {
        int c = e >> 7, i = e & 127;
        int idx = idx_s[i];
        float v  = emb[idx * C_DIM + c];
        float xv = xb[(size_t)c * HW + i];
        float d  = v - xv;
        lacc += d * d;
        out_q[((size_t)(b * C_DIM + c)) * HW + hw0 + i] = v;
    }
    red_s[tid] = lacc;
    __syncthreads();
    for (int s = THREADS / 2; s > 0; s >>= 1) {
        if (tid < s) red_s[tid] += red_s[tid + s];
        __syncthreads();
    }
    if (tid == 0) partials[bb] = red_s[0];
}

// ---------------------------------------------------------------------------
// Kernel 3: deterministic final reductions (loss, usage).
// ---------------------------------------------------------------------------
__global__ void vq_final(const float* __restrict__ partials,
                         const float* __restrict__ used,
                         float* __restrict__ out_loss,
                         float* __restrict__ out_usage) {
    __shared__ float s1[256], s2[256];
    int t = threadIdx.x;
    s1[t] = partials[t];
    s2[t] = used[t] + used[t + 256];
    __syncthreads();
    for (int s = 128; s > 0; s >>= 1) {
        if (t < s) { s1[t] += s1[t + s]; s2[t] += s2[t + s]; }
        __syncthreads();
    }
    if (t == 0) {
        // q_latent + 0.25 * e_latent, both equal mean((q - x)^2) in value
        out_loss[0]  = 1.25f * s1[0] / (float)Q_ELEMS;
        out_usage[0] = s2[0] / (float)K_CODES;
    }
}

// ---------------------------------------------------------------------------
extern "C" void kernel_launch(void* const* d_in, const int* in_sizes, int n_in,
                              void* d_out, int out_size, void* d_ws, size_t ws_size,
                              hipStream_t stream) {
    (void)in_sizes; (void)n_in; (void)out_size; (void)ws_size;
    const float* x   = (const float*)d_in[0];   // [32,128,32,32]
    const float* emb = (const float*)d_in[1];   // [512,128]

    char* ws = (char*)d_ws;
    unsigned short* ehi   = (unsigned short*)(ws);            // 128 KB
    unsigned short* elo   = (unsigned short*)(ws + 131072);   // 128 KB
    float*          enorm = (float*)(ws + 262144);            // 2 KB
    float*          used  = (float*)(ws + 264192);            // 2 KB
    float*          parts = (float*)(ws + 266240);            // 1 KB

    float* out       = (float*)d_out;
    float* out_loss  = out + Q_ELEMS;
    float* out_idx   = out + Q_ELEMS + 1;
    float* out_usage = out + Q_ELEMS + 1 + N_TOTAL;

    vq_prep <<<256, THREADS, 0, stream>>>(emb, ehi, elo, enorm, used);
    vq_main <<<N_TOTAL / COLS_PER_BLOCK, THREADS, 0, stream>>>(
        x, emb, ehi, elo, enorm, used, parts, out, out_idx);
    vq_final<<<1, THREADS, 0, stream>>>(parts, used, out_loss, out_usage);
}
